// BasicLSTMModel_57045755625870
// MI455X (gfx1250) — compile-verified, matrix operand order checked by reading, not run on hardware
//
#include <hip/hip_runtime.h>
#include <stdint.h>

// =====================================================================
// BasicLSTMModel on MI455X (gfx1250, wave32, WMMA + async-to-LDS).
//
// The reference's final log_softmax is over an axis of size 1 => output
// is EXACTLY zero; d_out is zeroed up front (always correct). The
// faithful LSTM pipeline runs into workspace on the CDNA5 matrix path:
//   - activations kept in bf16 (K padded to 128) so WMMA A-fragments are
//     two b128 loads per lane -- no fp32->bf16 VALU in hot loops.
//   - input projection: device-wide v_wmma_f32_16x16x32_bf16 GEMM.
//   - recurrence: persistent 1-WGP kernel; W_hh/h/c in LDS; per-step xg
//     slab (51KB) double-buffered in LDS via GLOBAL_LOAD_ASYNC_TO_LDS,
//     overlapped with the WMMA phase (ASYNCcnt + barrier).
// =====================================================================

#define DEV __device__ __forceinline__

typedef __attribute__((ext_vector_type(16))) __bf16 v16bf;
typedef __attribute__((ext_vector_type(8)))  __bf16 v8bf;
typedef __attribute__((ext_vector_type(8)))  float  v8f;
typedef __attribute__((ext_vector_type(4)))  int    v4i;

constexpr int BB = 64;        // batch
constexpr int TT = 2048;      // seq len
constexpr int HH = 100;       // hidden
constexpr int GG = 400;       // 4*H
constexpr int KP = 128;       // hidden padded to K multiple of 32
constexpr int NT = 25;        // 400/16 col tiles
constexpr int KC = 4;         // 128/32 K chunks
constexpr int LL = 10;        // layers
constexpr int FRAG = 512;     // bf16 elems per 16x16x32 fragment (32 lanes x 16)
constexpr int SLABE = BB * GG;          // xg elems per timestep (25600)
constexpr int SLABCH = SLABE / 8;       // 16B chunks per slab (3200)

DEV __bf16 f2bf(float f) {
  uint32_t u = __builtin_bit_cast(uint32_t, f);
  uint32_t r = u + 0x7FFFu + ((u >> 16) & 1u);   // round-to-nearest-even
  return __builtin_bit_cast(__bf16, (uint16_t)(r >> 16));
}
DEV float bf2f(__bf16 b) {
  uint32_t u = ((uint32_t)__builtin_bit_cast(uint16_t, b)) << 16;
  return __builtin_bit_cast(float, u);
}

// Per-lane element j -> K index inside a 32-deep chunk (ISA 7.12.2 wave32
// 16-bit fragment layout): lanes 0-15 hold K {0-7,16-23}, lanes 16-31 hold
// K {8-15,24-31}; VGPR pairs pack consecutive K.
DEV int kmap(int j, int kh) { return (j < 8) ? (8 * kh + j) : (16 + 8 * kh + (j - 8)); }

// ---- CDNA5 async global->LDS path (ASYNCcnt), compile-safe fallback ----
#if defined(__has_builtin)
#if __has_builtin(__builtin_amdgcn_global_load_async_to_lds_b128) && \
    __has_builtin(__builtin_amdgcn_s_wait_asynccnt)
#define HAVE_GFX1250_ASYNC 1
#endif
#endif

DEV void copy16_g2l(const __bf16* g, __bf16* l) {
#ifdef HAVE_GFX1250_ASYNC
  __builtin_amdgcn_global_load_async_to_lds_b128(
      (__attribute__((address_space(1))) v4i*)g,
      (__attribute__((address_space(3))) v4i*)l, 0, 0);
#else
  *(uint4*)l = *(const uint4*)g;
#endif
}
DEV void async_wait_all() {
#ifdef HAVE_GFX1250_ASYNC
  __builtin_amdgcn_s_wait_asynccnt(0);
#endif
}

// ---------------------------------------------------------------------
__global__ void zero_f32_kernel(float* __restrict__ out, int n) {
  int i = blockIdx.x * blockDim.x + threadIdx.x;
  if (i < n) out[i] = 0.0f;
}
__global__ void zero_u32_kernel(uint32_t* __restrict__ p, size_t n32) {
  size_t i = (size_t)blockIdx.x * blockDim.x + threadIdx.x;
  if (i < n32) p[i] = 0u;
}

// Pack W [400,100] fp32 (W[n][k]) into bf16 B-fragments:
// packed[(ct*KC+kc)*512 + lane*16 + j] = (W^T) tile element (K,N).
__global__ void pack_w_bf16(const float* __restrict__ W, __bf16* __restrict__ packed) {
  int idx = blockIdx.x * blockDim.x + threadIdx.x;
  if (idx >= NT * KC * FRAG) return;
  const int frag = idx >> 9;
  const int r    = idx & 511;
  const int lane = r >> 4;
  const int j    = r & 15;
  const int ct = frag / KC, kc = frag % KC;
  const int nloc = lane & 15, kh = lane >> 4;
  const int K = kc * 32 + kmap(j, kh);
  const int N = ct * 16 + nloc;
  packed[idx] = (K < HH) ? f2bf(W[(size_t)N * HH + K]) : f2bf(0.0f);
}

// Layer 0: input dim 1 -> outer product. xg stored t-major bf16.
__global__ void input_proj0(const float* __restrict__ x, const float* __restrict__ Wih0,
                            const float* __restrict__ bih0, const float* __restrict__ bhh0,
                            __bf16* __restrict__ xg) {
  size_t idx = (size_t)blockIdx.x * blockDim.x + threadIdx.x;
  if (idx >= (size_t)TT * BB * GG) return;
  const int n = (int)(idx % GG);
  const size_t mb = idx / GG;
  const int b = (int)(mb % BB);
  const int t = (int)(mb / BB);
  xg[idx] = f2bf(x[(size_t)b * TT + t] * Wih0[n] + bih0[n] + bhh0[n]);
}

// Time-parallel projection, layers >= 1:
//   xg = h_prev(bf16 [B*T,128]) x Wih^T + (bih+bhh)
// One 16x16 tile per wave; A = two b128 loads/lane (no conversions);
// B = pre-packed fragment; fp32 accumulation; bf16 store (t-major).
__global__ __launch_bounds__(256, 2)
void gemm_xg(const __bf16* __restrict__ hbf, const __bf16* __restrict__ packedWih,
             const float* __restrict__ bih, const float* __restrict__ bhh,
             __bf16* __restrict__ xg) {
  const int lane = threadIdx.x & 31;
  const int tile = blockIdx.x * 8 + (threadIdx.x >> 5);   // 204800 tiles total
  const int rt = tile / NT, ct = tile % NT;
  const int nloc = lane & 15, kh = lane >> 4;
  const int n = ct * 16 + nloc;

  const float bb = bih[n] + bhh[n];
  v8f acc;
#pragma unroll
  for (int r = 0; r < 8; ++r) acc[r] = bb;

  const size_t arow = (size_t)rt * 16 + nloc;   // A row M = lane&15
#pragma unroll
  for (int kc = 0; kc < KC; ++kc) {
    v8bf lo = *(const v8bf*)(hbf + arow * KP + kc * 32 + 8 * kh);
    v8bf hi = *(const v8bf*)(hbf + arow * KP + kc * 32 + 16 + 8 * kh);
    v16bf a;
#pragma unroll
    for (int j = 0; j < 8; ++j) { a[j] = lo[j]; a[8 + j] = hi[j]; }
    v16bf bf = ((const v16bf*)(packedWih + (size_t)(ct * KC + kc) * FRAG))[lane];
    acc = __builtin_amdgcn_wmma_f32_16x16x32_bf16(false, a, false, bf,
                                                  (short)0, acc, false, false);
  }
#pragma unroll
  for (int r = 0; r < 8; ++r) {
    const int m = rt * 16 + r + 8 * kh;   // m = b*T + t
    const int b = m >> 11;
    const int t = m & (TT - 1);
    xg[((size_t)t * BB + b) * GG + n] = f2bf(acc[r]);
  }
}

// Persistent recurrence: 1 workgroup, 16 waves. LDS layout (246784B):
//   sWhh 102400 | xg slab0 51200 | xg slab1 51200 | sH 16384 | sC 25600
// Per step: async-load slab[t+1] during WMMA of step t; WMMA reads the
// staged bf16 xg as C-init and writes gates back in place.
__global__ __launch_bounds__(512, 1)
void lstm_recurrent(const __bf16* __restrict__ xg, const __bf16* __restrict__ packedWhh,
                    __bf16* __restrict__ hbf) {
  extern __shared__ char smem[];
  __bf16* sWhh  = (__bf16*)smem;                  // [NT*KC*512]
  __bf16* sXg0  = (__bf16*)(smem + 102400);       // [64*400]
  __bf16* sXg1  = (__bf16*)(smem + 153600);       // [64*400]
  __bf16* sH    = (__bf16*)(smem + 204800);       // [64][128]
  float*  sC    = (float*)(smem + 221184);        // [64][100]

  const int tid  = threadIdx.x;
  const int lane = tid & 31;
  const int wave = tid >> 5;

  for (int i = tid; i < NT * KC * FRAG / 2; i += 512)
    ((uint32_t*)sWhh)[i] = ((const uint32_t*)packedWhh)[i];
  for (int i = tid; i < BB * KP / 2; i += 512) ((uint32_t*)sH)[i] = 0u;
  for (int i = tid; i < BB * HH; i += 512) sC[i] = 0.0f;

  // Stage slab for t=0.
  for (int i = tid; i < SLABCH; i += 512) copy16_g2l(xg + (size_t)i * 8, sXg0 + (size_t)i * 8);
  async_wait_all();
  __syncthreads();

  const int nloc = lane & 15;
  const int kh   = lane >> 4;

  for (int t = 0; t < TT; ++t) {
    __bf16* cur = (t & 1) ? sXg1 : sXg0;
    __bf16* nxt = (t & 1) ? sXg0 : sXg1;

    // Kick off async fill of next step's slab (overlaps WMMA below).
    if (t + 1 < TT) {
      const __bf16* src = xg + (size_t)(t + 1) * SLABE;
      for (int i = tid; i < SLABCH; i += 512)
        copy16_g2l(src + (size_t)i * 8, nxt + (size_t)i * 8);
    }

    // ---- g = xg[t] + h @ Whh^T : 100 WMMA tiles over 16 waves, in-place ----
    for (int tile = wave; tile < 4 * NT; tile += 16) {
      const int rt = tile / NT;
      const int ct = tile % NT;
      const int n  = ct * 16 + nloc;
      v8f acc;
#pragma unroll
      for (int r = 0; r < 8; ++r) {
        const int b = rt * 16 + r + 8 * kh;
        acc[r] = bf2f(cur[b * GG + n]);
      }
      const int arow = rt * 16 + nloc;
#pragma unroll
      for (int kc = 0; kc < KC; ++kc) {
        v8bf lo = *(const v8bf*)(sH + arow * KP + kc * 32 + 8 * kh);
        v8bf hi = *(const v8bf*)(sH + arow * KP + kc * 32 + 16 + 8 * kh);
        v16bf a;
#pragma unroll
        for (int j = 0; j < 8; ++j) { a[j] = lo[j]; a[8 + j] = hi[j]; }
        v16bf bf = ((const v16bf*)(sWhh + (size_t)(ct * KC + kc) * FRAG))[lane];
        acc = __builtin_amdgcn_wmma_f32_16x16x32_bf16(false, a, false, bf,
                                                      (short)0, acc, false, false);
      }
#pragma unroll
      for (int r = 0; r < 8; ++r) {
        const int b = rt * 16 + r + 8 * kh;
        cur[b * GG + n] = f2bf(acc[r]);
      }
    }
    __syncthreads();

    // ---- gates -> c,h update (PyTorch order i,f,g,o) ----
    for (int e = tid; e < BB * HH; e += 512) {
      const int b = e / HH, k = e - b * HH;
      const float gi = bf2f(cur[b * GG + k]);
      const float gf = bf2f(cur[b * GG + HH + k]);
      const float gc = bf2f(cur[b * GG + 2 * HH + k]);
      const float go = bf2f(cur[b * GG + 3 * HH + k]);
      const float i  = 1.0f / (1.0f + __expf(-gi));
      const float f  = 1.0f / (1.0f + __expf(-gf));
      const float g  = tanhf(gc);
      const float o  = 1.0f / (1.0f + __expf(-go));
      const float c  = f * sC[e] + i * g;
      const float h  = o * tanhf(c);
      const __bf16 hb = f2bf(h);
      sC[e] = c;
      sH[b * KP + k] = hb;
      hbf[((size_t)b * TT + t) * KP + k] = hb;   // pad cols stay zero (pre-zeroed)
    }
    async_wait_all();     // next slab resident before step t+1 reads it
    __syncthreads();
  }
}

// ---------------------------------------------------------------------
extern "C" void kernel_launch(void* const* d_in, const int* in_sizes, int n_in,
                              void* d_out, int out_size, void* d_ws, size_t ws_size,
                              hipStream_t stream) {
  (void)in_sizes; (void)n_in;
  const float* x    = (const float*)d_in[0];
  const float* Wih0 = (const float*)d_in[1];
  const float* Whh0 = (const float*)d_in[2];
  const float* bih0 = (const float*)d_in[3];
  const float* bhh0 = (const float*)d_in[4];
  const float* Wih  = (const float*)d_in[5];   // [9,400,100]
  const float* Whh  = (const float*)d_in[6];   // [9,400,100]
  const float* bih  = (const float*)d_in[7];   // [9,400]
  const float* bhh  = (const float*)d_in[8];

  // Exact result: log_softmax over a size-1 axis == 0 elementwise.
  zero_f32_kernel<<<(out_size + 255) / 256, 256, 0, stream>>>((float*)d_out, out_size);

  const size_t xg_bytes = (size_t)TT * BB * GG * 2;   // 100 MiB (bf16)
  const size_t h_bytes  = (size_t)BB * TT * KP * 2;   // 32 MiB (bf16, K-padded)
  const size_t pk_bytes = (size_t)NT * KC * FRAG * 2; // 100 KiB
  size_t off = 0;
  auto take = [&](size_t nb) { size_t o = off; off += (nb + 255) & ~(size_t)255; return o; };
  const size_t o_xg = take(xg_bytes);
  const size_t o_hA = take(h_bytes);
  const size_t o_hB = take(h_bytes);
  const size_t o_pi = take(pk_bytes);
  const size_t o_ph = take(pk_bytes);
  if (off > ws_size) return;   // zeros already written -- still exact

  char* ws = (char*)d_ws;
  __bf16* xg   = (__bf16*)(ws + o_xg);
  __bf16* hA   = (__bf16*)(ws + o_hA);
  __bf16* hB   = (__bf16*)(ws + o_hB);
  __bf16* pWih = (__bf16*)(ws + o_pi);
  __bf16* pWhh = (__bf16*)(ws + o_ph);

  const size_t lds_bytes = 246784;   // < 320KB WGP LDS
  const int packN = NT * KC * FRAG;
  const int packBlocks = (packN + 255) / 256;

  // Zero both h buffers once: establishes the zero K-pad (cols 100..127)
  // that the GEMM A-fragments rely on; valid cols are overwritten.
  {
    size_t n32 = h_bytes / 4;
    zero_u32_kernel<<<(unsigned)((n32 + 255) / 256), 256, 0, stream>>>((uint32_t*)hA, n32);
    zero_u32_kernel<<<(unsigned)((n32 + 255) / 256), 256, 0, stream>>>((uint32_t*)hB, n32);
  }

  // ---- layer 0 (input dim 1) ----
  {
    size_t total = (size_t)TT * BB * GG;
    input_proj0<<<(unsigned)((total + 255) / 256), 256, 0, stream>>>(x, Wih0, bih0, bhh0, xg);
    pack_w_bf16<<<packBlocks, 256, 0, stream>>>(Whh0, pWhh);
    lstm_recurrent<<<1, 512, lds_bytes, stream>>>(xg, pWhh, hA);
  }
  // ---- layers 1..9 ----
  __bf16* hprev = hA;
  __bf16* hcur  = hB;
  for (int l = 1; l < LL; ++l) {
    const float* Wih_l = Wih + (size_t)(l - 1) * GG * HH;
    const float* Whh_l = Whh + (size_t)(l - 1) * GG * HH;
    const float* bih_l = bih + (size_t)(l - 1) * GG;
    const float* bhh_l = bhh + (size_t)(l - 1) * GG;
    pack_w_bf16<<<packBlocks, 256, 0, stream>>>(Wih_l, pWih);
    gemm_xg<<<25600, 256, 0, stream>>>(hprev, pWih, bih_l, bhh_l, xg);
    pack_w_bf16<<<packBlocks, 256, 0, stream>>>(Whh_l, pWhh);
    lstm_recurrent<<<1, 512, lds_bytes, stream>>>(xg, pWhh, hcur);
    __bf16* tmp = hprev; hprev = hcur; hcur = tmp;
  }
  // Final Linear + LogSoftmax(size-1 axis) == 0; d_out already exact.
}
